// GroupedQueryAttention_66159676228272
// MI455X (gfx1250) — matrix operand order repaired and checked by reading, not compiled, not run on hardware
//
#include <hip/hip_runtime.h>
#include <hip/hip_bf16.h>

// ---------------------------------------------------------------------------
// Types for WMMA fragments (wave32, CDNA5 16x16x32 f16 WMMA)
// ---------------------------------------------------------------------------
typedef _Float16 half8  __attribute__((ext_vector_type(8)));
typedef _Float16 half16 __attribute__((ext_vector_type(16)));
typedef float    floatx8 __attribute__((ext_vector_type(8)));
typedef float    floatx4 __attribute__((ext_vector_type(4)));

static __device__ __forceinline__ half16 cat8(half8 lo, half8 hi) {
    return __builtin_shufflevector(lo, hi, 0,1,2,3,4,5,6,7,8,9,10,11,12,13,14,15);
}
static __device__ __forceinline__ floatx8 wmma_f16(half16 a, half16 b, floatx8 c) {
    // D = A(16x32 f16) * B(32x16 f16) + C(16x16 f32)  -> v_wmma_f32_16x16x32_f16
    return __builtin_amdgcn_wmma_f32_16x16x32_f16(false, a, false, b, (short)0, c, false, false);
}
static __device__ __forceinline__ floatx8 zero8() {
    floatx8 z;
#pragma unroll
    for (int i = 0; i < 8; ++i) z[i] = 0.0f;
    return z;
}
// Single v_rcp_f32 (~1 ulp) instead of IEEE divide chain
static __device__ __forceinline__ float fast_rcp(float x) {
    return __builtin_amdgcn_rcpf(x);
}

// Problem constants (fixed by the reference setup)
#define Bsz  2
#define SQ   2048
#define SKV  2048
#define DIN  2048
#define HQ   16
#define HKV  8
#define DH   128
#define LDP  40   // padded LDS row stride in halves for 32-wide K tiles

// ---------------------------------------------------------------------------
// Generic 128x128-tile GEMM:  C[M,N] = A[M,K] * Bw[K,N]
//   A_HALF: A is f16 (row-major) else f32 (row-major). Bw always f32.
//   OUT_MODE 0: store f16 row-major (ld = N)
//   OUT_MODE 1: store f32 row-major (ld = N)
//   OUT_MODE 2: scatter f16 to (b, h, s, d) layout  (row=b*SQ+s, col=h*DH+d)
// Block: 256 threads = 8 waves, wave grid 2x4, each wave does 64x32.
// ---------------------------------------------------------------------------
template <bool A_HALF, int OUT_MODE>
__global__ void __launch_bounds__(256)
gemm128_wmma(const void* __restrict__ Ap, const float* __restrict__ Bw,
             void* __restrict__ outp, int M, int N, int K) {
    __shared__ _Float16 As[128 * LDP];   // [m][k], k in 0..31
    __shared__ _Float16 Bs[128 * LDP];   // transposed: [n][k]

    const int tid  = threadIdx.x;
    const int lane = tid & 31;
    const int wave = tid >> 5;
    const int hl   = lane >> 4;      // which half of wave (0/1)
    const int ll   = lane & 15;      // lane within half
    const int bm   = blockIdx.y * 128;
    const int bn   = blockIdx.x * 128;
    const int wr   = wave >> 2;      // 0..1  (64 rows each)
    const int wc   = wave & 3;       // 0..3  (32 cols each)

    floatx8 acc[4][2];
#pragma unroll
    for (int i = 0; i < 4; ++i)
#pragma unroll
        for (int j = 0; j < 2; ++j) acc[i][j] = zero8();

    // staging assignments
    const int am     = tid >> 1;           // A row 0..127
    const int ak     = (tid & 1) * 16;     // A col 0 or 16
    const int bk     = tid >> 3;           // B row 0..31
    const int bn_off = (tid & 7) * 16;     // B col 0..112

    for (int k0 = 0; k0 < K; k0 += 32) {
        // ---- stage A tile (f32->f16 or f16 copy) ----
        if constexpr (A_HALF) {
            const _Float16* src = (const _Float16*)Ap + (size_t)(bm + am) * K + k0 + ak;
            *(half8*)&As[am * LDP + ak]     = *(const half8*)src;
            *(half8*)&As[am * LDP + ak + 8] = *(const half8*)(src + 8);
        } else {
            const float* src = (const float*)Ap + (size_t)(bm + am) * K + k0 + ak;
#pragma unroll
            for (int i = 0; i < 16; i += 4) {
                floatx4 v = *(const floatx4*)(src + i);
                As[am * LDP + ak + i + 0] = (_Float16)v.x;
                As[am * LDP + ak + i + 1] = (_Float16)v.y;
                As[am * LDP + ak + i + 2] = (_Float16)v.z;
                As[am * LDP + ak + i + 3] = (_Float16)v.w;
            }
        }
        // ---- stage B tile transposed: Bs[n][k] ----
        {
            const float* src = Bw + (size_t)(k0 + bk) * N + bn + bn_off;
#pragma unroll
            for (int i = 0; i < 16; i += 4) {
                floatx4 v = *(const floatx4*)(src + i);
                Bs[(bn_off + i + 0) * LDP + bk] = (_Float16)v.x;
                Bs[(bn_off + i + 1) * LDP + bk] = (_Float16)v.y;
                Bs[(bn_off + i + 2) * LDP + bk] = (_Float16)v.z;
                Bs[(bn_off + i + 3) * LDP + bk] = (_Float16)v.w;
            }
        }
        __syncthreads();

        // ---- fragments + WMMA ----
        half16 afr[4];
#pragma unroll
        for (int mt = 0; mt < 4; ++mt) {
            const _Float16* p = &As[(wr * 64 + mt * 16 + ll) * LDP + 8 * hl];
            afr[mt] = cat8(*(const half8*)p, *(const half8*)(p + 16));
        }
        half16 bfr[2];
#pragma unroll
        for (int nt = 0; nt < 2; ++nt) {
            const _Float16* p = &Bs[(wc * 32 + nt * 16 + ll) * LDP + 8 * hl];
            bfr[nt] = cat8(*(const half8*)p, *(const half8*)(p + 16));
        }
#pragma unroll
        for (int mt = 0; mt < 4; ++mt)
#pragma unroll
            for (int nt = 0; nt < 2; ++nt)
                acc[mt][nt] = wmma_f16(afr[mt], bfr[nt], acc[mt][nt]);
        __syncthreads();
    }

    // ---- epilogue ----
#pragma unroll
    for (int mt = 0; mt < 4; ++mt)
#pragma unroll
        for (int nt = 0; nt < 2; ++nt)
#pragma unroll
            for (int r = 0; r < 8; ++r) {
                const int gm = bm + wr * 64 + mt * 16 + r + 8 * hl;
                const int gn = bn + wc * 32 + nt * 16 + ll;
                const float v = acc[mt][nt][r];
                if constexpr (OUT_MODE == 0) {
                    ((_Float16*)outp)[(size_t)gm * N + gn] = (_Float16)v;
                } else if constexpr (OUT_MODE == 1) {
                    ((float*)outp)[(size_t)gm * N + gn] = v;
                } else {
                    const int b = gm >> 11, s = gm & 2047;
                    const int h = gn >> 7,  d = gn & 127;
                    ((_Float16*)outp)[(((size_t)(b * HQ + h) * SQ + s) << 7) + d] = (_Float16)v;
                }
            }
}

// ---------------------------------------------------------------------------
// RoPE on Q in place.  Qh layout: (b, h, s, d) f16; one thread per (row, j) pair.
// theta_j = 10^-j, angle = (s+1)*theta_j
// ---------------------------------------------------------------------------
__global__ void rope_q_kernel(_Float16* __restrict__ Qh, int total) {
    int idx = blockIdx.x * blockDim.x + threadIdx.x;
    if (idx >= total) return;
    const int j   = idx & 63;
    const int row = idx >> 6;            // b*HQ*SQ + h*SQ + s
    const int s   = row & 2047;
    const size_t base = ((size_t)row << 7) + 2 * j;
    const float ang = (float)(s + 1) * exp10f(-(float)j);
    float sn, cs;
    __sincosf(ang, &sn, &cs);
    const float xe = (float)Qh[base], xo = (float)Qh[base + 1];
    Qh[base]     = (_Float16)(xe * cs - xo * sn);
    Qh[base + 1] = (_Float16)(xe * sn + xo * cs);
}

// ---------------------------------------------------------------------------
// K: repeat(2, last axis) + rope, fused.  With xe==xo==v:
//   out[2j] = v*(cos-sin), out[2j+1] = v*(sin+cos),  v = Kraw[b, s, h*64 + j]
// Writes Kh in (b, h, s, d) f16.
// ---------------------------------------------------------------------------
__global__ void rope_repeat_k_kernel(const _Float16* __restrict__ Kraw,
                                     _Float16* __restrict__ Kh, int total) {
    int idx = blockIdx.x * blockDim.x + threadIdx.x;
    if (idx >= total) return;
    const int j   = idx & 63;
    const int row = idx >> 6;
    const int s   = row & 2047;
    const int bh  = row >> 11;
    const int b   = bh >> 4, h = bh & 15;
    const float v = (float)Kraw[((size_t)(b * SKV + s)) * (HKV * DH) + h * 64 + j];
    const float ang = (float)(s + 1) * exp10f(-(float)j);
    float sn, cs;
    __sincosf(ang, &sn, &cs);
    const size_t base = ((size_t)row << 7) + 2 * j;
    Kh[base]     = (_Float16)(v * (cs - sn));
    Kh[base + 1] = (_Float16)(v * (sn + cs));
}

// ---------------------------------------------------------------------------
// V: repeat(2, last axis), stored transposed (b, h, d, s) so P*V B-fragments
// are contiguous-in-kv loads.
// ---------------------------------------------------------------------------
__global__ void repeat_v_kernel(const _Float16* __restrict__ Vraw,
                                _Float16* __restrict__ Vt, int total) {
    int idx = blockIdx.x * blockDim.x + threadIdx.x;
    if (idx >= total) return;
    const int s  = idx & 2047;
    const int d  = (idx >> 11) & 127;
    const int bh = idx >> 18;
    const int b  = bh >> 4, h = bh & 15;
    Vt[idx] = Vraw[((size_t)(b * SKV + s)) * (HKV * DH) + h * 64 + (d >> 1)];
}

// ---------------------------------------------------------------------------
// Flash attention: one wave per 16-row Q tile.  Block = 128 threads = 4 waves.
// grid = (SQ/64, B*HQ).
// scores(16x32 per step) via 8 WMMAs, online softmax with shfl_xor row
// reductions, P transposed through wave-private LDS, O += P*V via 8 WMMAs.
// ---------------------------------------------------------------------------
__global__ void __launch_bounds__(128)
flash_attn_kernel(const _Float16* __restrict__ Qh, const _Float16* __restrict__ Kh,
                  const _Float16* __restrict__ Vt, const float* __restrict__ mask,
                  _Float16* __restrict__ attn) {
    __shared__ _Float16 Pl[4][16 * LDP];

    const int lane = threadIdx.x & 31;
    const int wave = threadIdx.x >> 5;
    const int hl   = lane >> 4, ll = lane & 15;
    const int bh   = blockIdx.y;            // 0..31
    const int b    = bh >> 4, h = bh & 15;
    const int q0   = blockIdx.x * 64 + wave * 16;

    const size_t headoff = (size_t)bh * SQ * DH;
    const _Float16* Qb = Qh + headoff;
    const _Float16* Kb = Kh + headoff;
    const _Float16* Vb = Vt + headoff;      // (d, s) within head

    // Q A-fragments for the whole 16x128 tile (4 k-chunks of 32)
    half16 qfr[4];
    {
        const _Float16* qrow = Qb + (size_t)(q0 + ll) * DH;
#pragma unroll
        for (int c = 0; c < 4; ++c) {
            const _Float16* p = qrow + 32 * c + 8 * hl;
            qfr[c] = cat8(*(const half8*)p, *(const half8*)(p + 16));
        }
    }

    floatx8 o[8];
#pragma unroll
    for (int j = 0; j < 8; ++j) o[j] = zero8();
    float mrow[8], lrow[8];
#pragma unroll
    for (int r = 0; r < 8; ++r) { mrow[r] = -3.0e38f; lrow[r] = 0.0f; }

    const float scale = 0.35355339059327373f;   // (D/Hq)^-0.5 = 8^-0.5

    for (int kv0 = 0; kv0 < SKV; kv0 += 32) {
        if (kv0 + 32 < SKV) {   // hint next K/V tiles toward the caches
            __builtin_prefetch(Kb + (size_t)(kv0 + 32 + ll) * DH, 0, 0);
            __builtin_prefetch(Vb + (size_t)(16 * ll) * SKV + kv0 + 32, 0, 0);
        }
        // ---- scores: two 16x16 tiles over kv columns (chunk-outer order) ----
        floatx8 s0 = zero8(), s1 = zero8();
        const _Float16* krow0 = Kb + (size_t)(kv0 + ll) * DH;        // tile t=0
        const _Float16* krow1 = Kb + (size_t)(kv0 + 16 + ll) * DH;   // tile t=1
#pragma unroll
        for (int c = 0; c < 4; ++c) {
            const _Float16* p0 = krow0 + 32 * c + 8 * hl;
            const _Float16* p1 = krow1 + 32 * c + 8 * hl;
            half16 b0 = cat8(*(const half8*)p0, *(const half8*)(p0 + 16));
            half16 b1 = cat8(*(const half8*)p1, *(const half8*)(p1 + 16));
            s0 = wmma_f16(qfr[c], b0, s0);
            s1 = wmma_f16(qfr[c], b1, s1);
        }
        // ---- scale + mask penalty + online softmax ----
#pragma unroll
        for (int r = 0; r < 8; ++r) {
            const int qrow = q0 + r + 8 * hl;
            const size_t mbase = ((size_t)b * SQ + qrow) * SKV + kv0;
            const float mk0 = mask[mbase + ll];
            const float mk1 = mask[mbase + 16 + ll];
            float a0 = s0[r] * scale - (fast_rcp(mk0) - 1.0f);
            float a1 = s1[r] * scale - (fast_rcp(mk1) - 1.0f);
            float rm = fmaxf(a0, a1);
#pragma unroll
            for (int off = 8; off; off >>= 1) rm = fmaxf(rm, __shfl_xor(rm, off, 32));
            const float newm = fmaxf(mrow[r], rm);
            const float corr = __expf(mrow[r] - newm);
            const float p0 = __expf(a0 - newm);
            const float p1 = __expf(a1 - newm);
            float rs = p0 + p1;
#pragma unroll
            for (int off = 8; off; off >>= 1) rs += __shfl_xor(rs, off, 32);
            lrow[r] = lrow[r] * corr + rs;
            mrow[r] = newm;
#pragma unroll
            for (int j = 0; j < 8; ++j) o[j][r] *= corr;
            const int ql = r + 8 * hl;
            Pl[wave][ql * LDP + ll]      = (_Float16)p0;
            Pl[wave][ql * LDP + 16 + ll] = (_Float16)p1;
        }
        // ---- P (C-layout) -> A-fragment via LDS round trip ----
        half16 pfr;
        {
            const _Float16* p = &Pl[wave][ll * LDP + 8 * hl];
            pfr = cat8(*(const half8*)p, *(const half8*)(p + 16));
        }
        // ---- O += P * V ----
#pragma unroll
        for (int j = 0; j < 8; ++j) {
            const _Float16* vp = Vb + (size_t)(16 * j + ll) * SKV + kv0 + 8 * hl;
            half16 vfr = cat8(*(const half8*)vp, *(const half8*)(vp + 16));
            o[j] = wmma_f16(pfr, vfr, o[j]);
        }
    }

    // ---- normalize and store to attn (b, s, h*128+d) as f16 ----
#pragma unroll
    for (int r = 0; r < 8; ++r) {
        const float inv = fast_rcp(lrow[r]);
        const int s = q0 + r + 8 * hl;
#pragma unroll
        for (int j = 0; j < 8; ++j)
            attn[((size_t)b * SQ + s) * (HQ * DH) + h * DH + 16 * j + ll] =
                (_Float16)(o[j][r] * inv);
    }
}

// ---------------------------------------------------------------------------
// Launcher
// ---------------------------------------------------------------------------
extern "C" void kernel_launch(void* const* d_in, const int* in_sizes, int n_in,
                              void* d_out, int out_size, void* d_ws, size_t ws_size,
                              hipStream_t stream) {
    (void)in_sizes; (void)n_in; (void)out_size; (void)ws_size;
    const float* q    = (const float*)d_in[0];
    const float* kv   = (const float*)d_in[1];
    const float* mask = (const float*)d_in[2];
    const float* Wq   = (const float*)d_in[3];
    const float* Wk   = (const float*)d_in[4];
    const float* Wv   = (const float*)d_in[5];
    const float* Wo   = (const float*)d_in[6];
    float* out = (float*)d_out;

    char* ws = (char*)d_ws;
    _Float16* Qh   = (_Float16*)(ws + (size_t)0);               // 16 MB (b,h,s,d)
    _Float16* Kh   = (_Float16*)(ws + ((size_t)16 << 20));      // 16 MB (b,h,s,d)
    _Float16* Vt   = (_Float16*)(ws + ((size_t)32 << 20));      // 16 MB (b,h,d,s)
    _Float16* attn = (_Float16*)(ws + ((size_t)48 << 20));      // 16 MB (b,s,h*d)
    _Float16* Kraw = (_Float16*)(ws + ((size_t)64 << 20));      //  8 MB (b,s,1024)
    _Float16* Vraw = (_Float16*)(ws + ((size_t)72 << 20));      //  8 MB (b,s,1024)

    // 1) Projections (f32 A, f32 W -> f16 out via WMMA)
    gemm128_wmma<false, 2><<<dim3(2048 / 128, 4096 / 128), 256, 0, stream>>>(
        q, Wq, Qh, Bsz * SQ, HQ * DH, DIN);
    gemm128_wmma<false, 0><<<dim3(1024 / 128, 4096 / 128), 256, 0, stream>>>(
        kv, Wk, Kraw, Bsz * SKV, HKV * DH, DIN);
    gemm128_wmma<false, 0><<<dim3(1024 / 128, 4096 / 128), 256, 0, stream>>>(
        kv, Wv, Vraw, Bsz * SKV, HKV * DH, DIN);

    // 2) RoPE / repeat expansions
    const int nq = Bsz * HQ * SQ * 64;
    rope_q_kernel<<<nq / 256, 256, 0, stream>>>(Qh, nq);
    rope_repeat_k_kernel<<<nq / 256, 256, 0, stream>>>(Kraw, Kh, nq);
    const int nv = Bsz * HQ * SKV * DH;
    repeat_v_kernel<<<nv / 256, 256, 0, stream>>>(Vraw, Vt, nv);

    // 3) Attention (flash-style, WMMA for QK^T and P*V)
    flash_attn_kernel<<<dim3(SQ / 64, Bsz * HQ), 128, 0, stream>>>(Qh, Kh, Vt, mask, attn);

    // 4) Output projection (f16 A, f32 W -> f32 out)
    gemm128_wmma<true, 1><<<dim3(2048 / 128, 4096 / 128), 256, 0, stream>>>(
        attn, Wo, out, Bsz * SQ, DIN, HQ * DH);
}
